// KeOpsMultiHeadLatentAttention_79259326480573
// MI455X (gfx1250) — matrix operand order repaired and checked by reading, not compiled, not run on hardware
//
#include <hip/hip_runtime.h>
#include <hip/hip_bf16.h>

// ---------------------------------------------------------------------------
// MLA forward for MI455X (gfx1250).
// All hot GEMM/attention loops run on v_wmma_f32_16x16x32_bf16 with bf16
// operands pre-staged in memory, so inner loops are b128 loads + WMMA only.
// ---------------------------------------------------------------------------

typedef __attribute__((ext_vector_type(16))) __bf16 v16bf;
typedef __attribute__((ext_vector_type(8)))  __bf16 v8bf;
typedef __attribute__((ext_vector_type(4)))  __bf16 v4bf;
typedef __attribute__((ext_vector_type(8)))  float  v8f;

#define DIM_C    1024
#define LATENT_C 512
#define HEADS_C  16
#define HD_C     64
#define SCALE_C  0.125f      // HEAD_DIM^-0.5

__device__ __forceinline__ v8f wmma_bf16(v16bf a, v16bf b, v8f c) {
  return __builtin_amdgcn_wmma_f32_16x16x32_bf16(
      false, a, false, b, (short)0, c, false, false);
}

// A-fragment (16x32 MxK) from row-major bf16, leading dim lda.
// ISA layout: lane l -> row M=l&15, element e -> K=(e&7)+16*(e>>3)+8*(l>>4).
// Per lane that is two contiguous 8-element runs -> 2 x global_load_b128.
__device__ __forceinline__ v16bf load_a_frag(const __bf16* __restrict__ A,
                                             int lda, int row0, int k0,
                                             int lane) {
  const int m = lane & 15, half = lane >> 4;
  const __bf16* p = A + (size_t)(row0 + m) * lda + k0 + 8 * half;
  v8bf lo = *(const v8bf*)(p);        // K = 8*half + 0..7
  v8bf hi = *(const v8bf*)(p + 16);   // K = 16 + 8*half + 0..7
  v16bf a;
#pragma unroll
  for (int e = 0; e < 8; ++e) { a[e] = lo[e]; a[8 + e] = hi[e]; }
  return a;
}

// B-fragment (32x16 KxN) from Bt = B^T stored row-major (N x K), leading dim ldb.
// ISA layout: lane l -> col N=l&15, element e -> K=(e&7)+16*(e>>3)+8*(l>>4).
// Per lane that is two contiguous 8-element runs -> 2 x global_load_b128.
__device__ __forceinline__ v16bf load_b_frag(const __bf16* __restrict__ Bt,
                                             int ldb, int k0, int col0,
                                             int lane) {
  const int nn = lane & 15, half = lane >> 4;
  const __bf16* p = Bt + (size_t)(col0 + nn) * ldb + k0 + 8 * half;
  v8bf lo = *(const v8bf*)(p);
  v8bf hi = *(const v8bf*)(p + 16);
  v16bf bv;
#pragma unroll
  for (int e = 0; e < 8; ++e) { bv[e] = lo[e]; bv[8 + e] = hi[e]; }
  return bv;
}

// ---------------------------------------------------------------------------
// Prep kernels (one-time, outside hot path)
// ---------------------------------------------------------------------------
__global__ __launch_bounds__(256)
void cvt_f32_bf16_kernel(const float* __restrict__ s, __bf16* __restrict__ d,
                         int n4) {
  const int i = blockIdx.x * 256 + threadIdx.x;
  if (i < n4) {
    const float4 f = ((const float4*)s)[i];
    v4bf o = { (__bf16)f.x, (__bf16)f.y, (__bf16)f.z, (__bf16)f.w };
    ((v4bf*)d)[i] = o;
  }
}

// src: R x C f32 -> dst: C x R bf16 (tiled transpose through LDS)
__global__ __launch_bounds__(256)
void transpose_f32_bf16_kernel(const float* __restrict__ src,
                               __bf16* __restrict__ dst, int R, int C) {
  __shared__ float tile[32][33];
  const int tx = threadIdx.x & 31, ty = threadIdx.x >> 5;   // ty: 0..7
  const int r0 = blockIdx.y * 32, c0 = blockIdx.x * 32;
#pragma unroll
  for (int i = 0; i < 4; ++i)
    tile[ty + 8 * i][tx] = src[(size_t)(r0 + ty + 8 * i) * C + c0 + tx];
  __syncthreads();
#pragma unroll
  for (int i = 0; i < 4; ++i)
    dst[(size_t)(c0 + ty + 8 * i) * R + r0 + tx] = (__bf16)tile[tx][ty + 8 * i];
}

// src: R x C bf16 -> dst: C x R bf16
__global__ __launch_bounds__(256)
void transpose_bf16_kernel(const __bf16* __restrict__ src,
                           __bf16* __restrict__ dst, int R, int C) {
  __shared__ __bf16 tile[32][34];
  const int tx = threadIdx.x & 31, ty = threadIdx.x >> 5;
  const int r0 = blockIdx.y * 32, c0 = blockIdx.x * 32;
#pragma unroll
  for (int i = 0; i < 4; ++i)
    tile[ty + 8 * i][tx] = src[(size_t)(r0 + ty + 8 * i) * C + c0 + tx];
  __syncthreads();
#pragma unroll
  for (int i = 0; i < 4; ++i)
    dst[(size_t)(c0 + ty + 8 * i) * R + r0 + tx] = tile[tx][ty + 8 * i];
}

// ---------------------------------------------------------------------------
// GEMM: C[M x N] = act(A[M x K] @ B[K x N] + bias), B given as B^T (N x K).
// Block = 128 threads = 4 waves; each wave computes 32x64 (2 A-frags reuse
// 4 B-frags -> 8 WMMA per 12 b128 loads per K-step). Dims: M%128==0,
// N%64==0, K%32==0 at every call site.
// ---------------------------------------------------------------------------
__global__ __launch_bounds__(128)
void gemm_bf16_kernel(const __bf16* __restrict__ A, int lda,
                      const __bf16* __restrict__ Bt, int ldb,
                      void* __restrict__ Cout, int ldc,
                      const float* __restrict__ bias,
                      int M, int N, int K, int do_relu, int out_f32) {
  const int lane = threadIdx.x & 31;
  const int wave = threadIdx.x >> 5;               // 0..3
  const int m0   = blockIdx.y * 128 + wave * 32;
  const int n0   = blockIdx.x * 64;

  v8f acc[2][4] = {};

  for (int k0 = 0; k0 < K; k0 += 32) {
    if (k0 + 32 < K)   // hint next A tile (global_prefetch_b8)
      __builtin_prefetch(A + (size_t)(m0 + (lane & 15)) * lda + k0 + 32, 0, 3);

    v16bf a0 = load_a_frag(A, lda, m0,      k0, lane);
    v16bf a1 = load_a_frag(A, lda, m0 + 16, k0, lane);
#pragma unroll
    for (int t = 0; t < 4; ++t) {
      v16bf b = load_b_frag(Bt, ldb, k0, n0 + t * 16, lane);
      acc[0][t] = wmma_bf16(a0, b, acc[0][t]);
      acc[1][t] = wmma_bf16(a1, b, acc[1][t]);
    }
  }

  const int nn = lane & 15, half = lane >> 4;
#pragma unroll
  for (int r = 0; r < 2; ++r) {
#pragma unroll
    for (int t = 0; t < 4; ++t) {
      const int col = n0 + t * 16 + nn;
      const float bv = bias ? bias[col] : 0.0f;
#pragma unroll
      for (int v = 0; v < 8; ++v) {
        const int row = m0 + r * 16 + v + 8 * half;
        float val = acc[r][t][v] + bv;
        if (do_relu) val = fmaxf(val, 0.0f);
        if (out_f32) ((float*)Cout)[(size_t)row * ldc + col] = val;
        else         ((__bf16*)Cout)[(size_t)row * ldc + col] = (__bf16)val;
      }
    }
  }
}

// ---------------------------------------------------------------------------
// Attention: one wave per (batch, head, 16-query tile). All operands bf16.
//   S = Q @ K^T            (K row-major == B-frag contiguous)
//   E = mask ? exp(S*scale) : 0   (reference: no max subtraction)
//   den += E @ ones ; num += E @ V   (V pre-transposed -> contiguous B-frags)
// E is re-laid C->A through bf16 LDS (stores b16, reads 2 x ds_load_b128).
// ---------------------------------------------------------------------------
__global__ __launch_bounds__(32)
void mla_attention_kernel(const __bf16* __restrict__ q,   // (B*N, DIM)
                          const __bf16* __restrict__ k,   // (B*N, DIM)
                          const __bf16* __restrict__ vT,  // (DIM, B*N)
                          const int*    __restrict__ mask,// (B, N)
                          __bf16* __restrict__ attn,      // (B*N, DIM)
                          int n, int mtot) {
  __shared__ __bf16 Els[16][40];                  // 80B row pitch, 16B aligned

  const int lane = threadIdx.x;                   // 0..31 (wave32)
  const int qt = blockIdx.x, h = blockIdx.y, b = blockIdx.z;
  const int row0 = b * n + qt * 16;
  const int col0 = h * HD_C;
  const int m = lane & 15, half = lane >> 4;

  v16bf qa[2];
#pragma unroll
  for (int s = 0; s < 2; ++s)
    qa[s] = load_a_frag(q, DIM_C, row0, col0 + s * 32, lane);

  v16bf onesb;
#pragma unroll
  for (int e = 0; e < 16; ++e) onesb[e] = (__bf16)1.0f;

  v8f num[4] = {};
  v8f den    = {};

  for (int kt = 0; kt < n; kt += 32) {
#pragma unroll
    for (int t = 0; t < 2; ++t) {
      v8f s8 = {};
#pragma unroll
      for (int s = 0; s < 2; ++s) {
        // B-frag of K^T: lane = key column, elements walk head-dim (contig)
        v16bf kb = load_b_frag(k, DIM_C, col0 + s * 32, b * n + kt + t * 16, lane);
        s8 = wmma_bf16(qa[s], kb, s8);
      }
      const int key   = kt + t * 16 + m;
      const bool keep = (mask[b * n + key] == 1);
#pragma unroll
      for (int vv = 0; vv < 8; ++vv)
        Els[vv + 8 * half][t * 16 + m] =
            keep ? (__bf16)__expf(s8[vv] * SCALE_C) : (__bf16)0.0f;
    }
    __syncthreads();

    v16bf ea;
    {
      v8bf lo = *(const v8bf*)&Els[m][8 * half];        // K = 8*half + 0..7
      v8bf hi = *(const v8bf*)&Els[m][16 + 8 * half];   // K = 16+8*half+0..7
#pragma unroll
      for (int e = 0; e < 8; ++e) { ea[e] = lo[e]; ea[8 + e] = hi[e]; }
    }
    __syncthreads();

    den = wmma_bf16(ea, onesb, den);                    // rowsum trick
#pragma unroll
    for (int t = 0; t < 4; ++t) {
      // B-frag of V: from vT, lane = head-dim col, elements walk keys (contig)
      v16bf vb = load_b_frag(vT, mtot, b * n + kt, col0 + t * 16, lane);
      num[t] = wmma_bf16(ea, vb, num[t]);
    }
  }

#pragma unroll
  for (int t = 0; t < 4; ++t)
#pragma unroll
    for (int vv = 0; vv < 8; ++vv) {
      const int row = row0 + vv + 8 * half;
      attn[(size_t)row * DIM_C + col0 + t * 16 + m] =
          (__bf16)(num[t][vv] / (den[vv] + 1e-6f));
    }
}

// ---------------------------------------------------------------------------
// Launch
// ---------------------------------------------------------------------------
extern "C" void kernel_launch(void* const* d_in, const int* in_sizes, int n_in,
                              void* d_out, int out_size, void* d_ws, size_t ws_size,
                              hipStream_t stream) {
  (void)in_sizes; (void)n_in; (void)out_size; (void)ws_size;

  const float* x    = (const float*)d_in[0];
  const int*   mask = (const int*)  d_in[1];
  const float* wq   = (const float*)d_in[2];
  const float* wkv  = (const float*)d_in[3];
  const float* lk1  = (const float*)d_in[4];
  const float* lk2  = (const float*)d_in[5];
  const float* lv1  = (const float*)d_in[6];
  const float* lv2  = (const float*)d_in[7];
  const float* wout = (const float*)d_in[8];
  const float* bout = (const float*)d_in[9];
  float* out        = (float*)d_out;

  const int B = 2, N = 2048;
  const int M = B * N;                          // 4096

  // Workspace layout (MB offsets), all bf16 unless noted:
  char* ws = (char*)d_ws;
  __bf16* xbf   = (__bf16*)(ws);                        //  0: 4096x1024 (8MB)
  __bf16* qbf   = (__bf16*)(ws + ((size_t) 8 << 20));   //  8: 4096x1024 (8MB)
  __bf16* kvbf  = (__bf16*)(ws + ((size_t)16 << 20));   // 16: 4096x2048 (16MB)
  __bf16* kbf   = (__bf16*)(ws + ((size_t)32 << 20));   // 32: 4096x1024 (8MB)
  __bf16* vbf   = (__bf16*)(ws + ((size_t)40 << 20));   // 40: 4096x1024 (8MB)
  __bf16* vTbf  = (__bf16*)(ws + ((size_t)48 << 20));   // 48: 1024x4096 (8MB)
  __bf16* tmpbf = (__bf16*)(ws + ((size_t)56 << 20));   // 56: 4096x512  (4MB)
  __bf16* attbf = (__bf16*)(ws + ((size_t)60 << 20));   // 60: 4096x1024 (8MB)
  __bf16* wqT   = (__bf16*)(ws + ((size_t)68 << 20));   // 68: 1024x1024 (2MB)
  __bf16* wkvT  = (__bf16*)(ws + ((size_t)70 << 20));   // 70: 2048x1024 (4MB)
  __bf16* lk1T  = (__bf16*)(ws + ((size_t)74 << 20));   // 74:  512x1024 (1MB)
  __bf16* lk2T  = (__bf16*)(ws + ((size_t)75 << 20));   // 75: 1024x512  (1MB)
  __bf16* lv1T  = (__bf16*)(ws + ((size_t)76 << 20));   // 76:  512x1024 (1MB)
  __bf16* lv2T  = (__bf16*)(ws + ((size_t)77 << 20));   // 77: 1024x512  (1MB)
  __bf16* woutT = (__bf16*)(ws + ((size_t)78 << 20));   // 78: 1024x1024 (2MB)

  // ---- prep: bf16 conversions / weight transposes ----
  {
    const int n4 = M * DIM_C / 4;
    cvt_f32_bf16_kernel<<<(n4 + 255) / 256, 256, 0, stream>>>(x, xbf, n4);
    auto tgrid = [](int R, int C) { return dim3(C / 32, R / 32); };
    transpose_f32_bf16_kernel<<<tgrid(DIM_C,    DIM_C),     256, 0, stream>>>(wq,   wqT,   DIM_C,    DIM_C);
    transpose_f32_bf16_kernel<<<tgrid(DIM_C,    2 * DIM_C), 256, 0, stream>>>(wkv,  wkvT,  DIM_C,    2 * DIM_C);
    transpose_f32_bf16_kernel<<<tgrid(DIM_C,    LATENT_C),  256, 0, stream>>>(lk1,  lk1T,  DIM_C,    LATENT_C);
    transpose_f32_bf16_kernel<<<tgrid(LATENT_C, DIM_C),     256, 0, stream>>>(lk2,  lk2T,  LATENT_C, DIM_C);
    transpose_f32_bf16_kernel<<<tgrid(DIM_C,    LATENT_C),  256, 0, stream>>>(lv1,  lv1T,  DIM_C,    LATENT_C);
    transpose_f32_bf16_kernel<<<tgrid(LATENT_C, DIM_C),     256, 0, stream>>>(lv2,  lv2T,  LATENT_C, DIM_C);
    transpose_f32_bf16_kernel<<<tgrid(DIM_C,    DIM_C),     256, 0, stream>>>(wout, woutT, DIM_C,    DIM_C);
  }

  const dim3 blk(128);
  auto grid = [](int n_, int m_) { return dim3(n_ / 64, m_ / 128); };

  // q = x @ wq
  gemm_bf16_kernel<<<grid(DIM_C, M), blk, 0, stream>>>(
      xbf, DIM_C, wqT, DIM_C, qbf, DIM_C, nullptr, M, DIM_C, DIM_C, 0, 0);
  // kv = x @ wkv
  gemm_bf16_kernel<<<grid(2 * DIM_C, M), blk, 0, stream>>>(
      xbf, DIM_C, wkvT, DIM_C, kvbf, 2 * DIM_C, nullptr, M, 2 * DIM_C, DIM_C, 0, 0);
  // k = (relu(kv[:, :DIM] @ lk1)) @ lk2
  gemm_bf16_kernel<<<grid(LATENT_C, M), blk, 0, stream>>>(
      kvbf, 2 * DIM_C, lk1T, DIM_C, tmpbf, LATENT_C, nullptr, M, LATENT_C, DIM_C, 1, 0);
  gemm_bf16_kernel<<<grid(DIM_C, M), blk, 0, stream>>>(
      tmpbf, LATENT_C, lk2T, LATENT_C, kbf, DIM_C, nullptr, M, DIM_C, LATENT_C, 0, 0);
  // v = (relu(kv[:, DIM:] @ lv1)) @ lv2
  gemm_bf16_kernel<<<grid(LATENT_C, M), blk, 0, stream>>>(
      kvbf + DIM_C, 2 * DIM_C, lv1T, DIM_C, tmpbf, LATENT_C, nullptr, M, LATENT_C, DIM_C, 1, 0);
  gemm_bf16_kernel<<<grid(DIM_C, M), blk, 0, stream>>>(
      tmpbf, LATENT_C, lv2T, LATENT_C, vbf, DIM_C, nullptr, M, DIM_C, LATENT_C, 0, 0);

  // vT for contiguous attention B-fragments
  transpose_bf16_kernel<<<dim3(DIM_C / 32, M / 32), 256, 0, stream>>>(
      vbf, vTbf, M, DIM_C);

  // attention
  mla_attention_kernel<<<dim3(N / 16, HEADS_C, B), dim3(32), 0, stream>>>(
      qbf, kbf, vTbf, mask, attbf, N, M);

  // out = attn @ wout + bout   (f32 result)
  gemm_bf16_kernel<<<grid(DIM_C, M), blk, 0, stream>>>(
      attbf, DIM_C, woutT, DIM_C, out, DIM_C, bout, M, DIM_C, DIM_C, 0, 1);
}
